// GRU_29042568856008
// MI455X (gfx1250) — matrix operand order repaired and checked by reading, not compiled, not run on hardware
//
#include <hip/hip_runtime.h>
#include <math.h>

#define T_STEPS 65536
#define IN_SZ   256
#define HID     512
#define NWG     16
#define HPW     (HID / NWG)   // 32 hidden units per workgroup
#define ROWS    (3 * HPW)     // 96 weight rows per workgroup (r,z,n gates)
#define WHH_LD  520           // padded LDS row stride (halves), 16B-aligned rows, bank-friendly
#define WIH_LD  264
#define BLOCK   256           // 8 waves of 32

typedef __bf16 bf16;
typedef __attribute__((ext_vector_type(16))) __bf16 v16bf;
typedef __attribute__((ext_vector_type(8)))  __bf16 v8bf;
typedef __attribute__((ext_vector_type(8)))  float  v8f;

union V16U { v16bf v; v8bf h[2]; };

__device__ __forceinline__ float sigmoid_(float x) { return 1.0f / (1.0f + __expf(-x)); }

// Split-precision matvec: rows [mt*16, mt*16+16) of W @ vec with W ~ (Whi + Wlo),
// vec ~ (vhi + vlo) as bf16 pairs.  acc += Ahi*Bhi + Ahi*Blo + Alo*Bhi  (~fp32 accuracy).
// B operand broadcasts vec into all 16 columns (columns of D are independent, so no
// masking needed); the result is read from column N=0 (lanes 0 and 16).
template <int KT>
__device__ __forceinline__ void matvec16s(const bf16* lwh, const bf16* lwl, int ld,
                                          const bf16* vh, const bf16* vl,
                                          int mt, int lane, float* dst) {
  v8f acc = {};
  const int khalf8 = (lane >> 4) * 8;                 // 0 or 8
  const int roff   = (mt * 16 + (lane & 15)) * ld + khalf8;
  const bf16* wh = lwh + roff;
  const bf16* wl = lwl + roff;
  const bf16* hh = vh + khalf8;                       // + khalf*8
  const bf16* hl = vl + khalf8;
#pragma unroll
  for (int kk = 0; kk < KT; ++kk) {
    V16U ah, al, bh, bl;
    ah.h[0] = *(const v8bf*)(wh + kk * 32);           // K = base+0..7
    ah.h[1] = *(const v8bf*)(wh + kk * 32 + 16);      // K = base+16..23
    al.h[0] = *(const v8bf*)(wl + kk * 32);
    al.h[1] = *(const v8bf*)(wl + kk * 32 + 16);
    bh.h[0] = *(const v8bf*)(hh + kk * 32);           // K = base+0..7, base+16..23 (broadcast)
    bh.h[1] = *(const v8bf*)(hh + kk * 32 + 16);
    bl.h[0] = *(const v8bf*)(hl + kk * 32);
    bl.h[1] = *(const v8bf*)(hl + kk * 32 + 16);
    acc = __builtin_amdgcn_wmma_f32_16x16x32_bf16(false, ah.v, false, bh.v,
                                                  (short)0, acc, false, false);
    acc = __builtin_amdgcn_wmma_f32_16x16x32_bf16(false, ah.v, false, bl.v,
                                                  (short)0, acc, false, false);
    acc = __builtin_amdgcn_wmma_f32_16x16x32_bf16(false, al.v, false, bh.v,
                                                  (short)0, acc, false, false);
  }
  if ((lane & 15) == 0) {                             // lanes 0 and 16 hold column N=0
    const int mb = mt * 16 + khalf8;                  // M = r (lane0) / r+8 (lane16)
#pragma unroll
    for (int r = 0; r < 8; ++r) dst[mb + r] = acc[r];
  }
}

__global__ void gru_init(unsigned* bar, float* gh) {
  if (threadIdx.x == 0) *bar = 0u;
  if (threadIdx.x < HID) gh[threadIdx.x] = 0.0f;
}

__global__ __launch_bounds__(BLOCK) void gru_persistent(
    const float* __restrict__ xs, const float* __restrict__ w_ih,
    const float* __restrict__ w_hh, const float* __restrict__ bias,
    const float* __restrict__ bias_n, float* __restrict__ out,
    float* gh, unsigned* bar) {
  extern __shared__ __align__(16) char smem[];
  bf16*  lds_whh_h = (bf16*)smem;                     // 96 x 520 halves
  bf16*  lds_whh_l = lds_whh_h + ROWS * WHH_LD;
  bf16*  lds_wih_h = lds_whh_l + ROWS * WHH_LD;       // 96 x 264 halves
  bf16*  lds_wih_l = lds_wih_h + ROWS * WIH_LD;
  bf16*  lds_hb_h  = lds_wih_l + ROWS * WIH_LD;       // 512 halves
  bf16*  lds_hb_l  = lds_hb_h + HID;
  bf16*  lds_xb_h  = lds_hb_l + HID;                  // 2 x 256 halves (double buffer)
  bf16*  lds_xb_l  = lds_xb_h + 2 * IN_SZ;
  float* lds_h     = (float*)(lds_xb_l + 2 * IN_SZ);  // 512 f32 (h exact)
  float* lds_hg    = lds_h + HID;                     // 96 f32 (W_hh @ h)
  float* lds_ig    = lds_hg + ROWS;                   // 2 x 96 f32 (W_ih @ x, pipelined)
  float* lds_bias  = lds_ig + 2 * ROWS;               // 96 f32
  float* lds_bn    = lds_bias + ROWS;                 // 32 f32

  const int tid  = threadIdx.x;
  const int lane = tid & 31;
  const int wave = tid >> 5;
  const int wg   = blockIdx.x;
  const int j0   = wg * HPW;

  // Are we running as a real workgroup cluster?  IB_STS2.CLUSTER_ID[9:6] != 0.
  unsigned cid;
  asm volatile("s_getreg_b32 %0, hwreg(HW_REG_IB_STS2, 6, 4)" : "=s"(cid));
  const bool in_cluster = (cid != 0);

  // ---- prologue: stage this WG's weight slice (hi/lo bf16 split) + biases ----
  for (int i = tid; i < ROWS * HID; i += BLOCK) {
    int lr = i / HID, k = i - lr * HID;
    int g = lr / HPW, jj = lr - g * HPW;
    float w = w_hh[(size_t)(g * HID + j0 + jj) * HID + k];
    bf16 whi = (bf16)w;
    lds_whh_h[lr * WHH_LD + k] = whi;
    lds_whh_l[lr * WHH_LD + k] = (bf16)(w - (float)whi);
  }
  for (int i = tid; i < ROWS * IN_SZ; i += BLOCK) {
    int lr = i / IN_SZ, k = i - lr * IN_SZ;
    int g = lr / HPW, jj = lr - g * HPW;
    float w = w_ih[(size_t)(g * HID + j0 + jj) * IN_SZ + k];
    bf16 whi = (bf16)w;
    lds_wih_h[lr * WIH_LD + k] = whi;
    lds_wih_l[lr * WIH_LD + k] = (bf16)(w - (float)whi);
  }
  for (int i = tid; i < ROWS; i += BLOCK) {
    int g = i / HPW, jj = i - g * HPW;
    lds_bias[i] = bias[g * HID + j0 + jj];
  }
  if (tid < HPW) lds_bn[tid] = bias_n[j0 + tid];
  for (int i = tid; i < HID; i += BLOCK) {
    lds_h[i] = 0.0f; lds_hb_h[i] = (bf16)0.0f; lds_hb_l[i] = (bf16)0.0f;
  }
  for (int i = tid; i < IN_SZ; i += BLOCK) {
    float x0 = xs[i], x1 = xs[IN_SZ + i];
    bf16 h0 = (bf16)x0, h1 = (bf16)x1;
    lds_xb_h[i] = h0;          lds_xb_l[i] = (bf16)(x0 - (float)h0);          // x_0
    lds_xb_h[IN_SZ + i] = h1;  lds_xb_l[IN_SZ + i] = (bf16)(x1 - (float)h1);  // x_1
  }
  __syncthreads();

  // ig_0 = W_ih @ x_0 (waves 6,7), into lds_ig parity 0
  if (wave >= 6) {
#pragma unroll
    for (int q = 0; q < 3; ++q)
      matvec16s<8>(lds_wih_h, lds_wih_l, WIH_LD, lds_xb_h, lds_xb_l,
                   (wave - 6) * 3 + q, lane, lds_ig);
  }

#pragma unroll 1
  for (int t = 0; t < T_STEPS; ++t) {
    __syncthreads();  // (A) lds_hb / lds_xb / lds_ig ready

    if (wave < 6) {
      // issue x_{t+2} global loads first so HBM/L2 latency hides under the WMMA chain
      const int tc = (t + 2 < T_STEPS) ? t + 2 : T_STEPS - 1;
      const float xv0 = xs[(size_t)tc * IN_SZ + tid];                    // tid in [0,192)
      const float xv1 = (tid < IN_SZ - 192) ? xs[(size_t)tc * IN_SZ + 192 + tid] : 0.0f;

      // hidden matvec for step t: rows [wave*16, wave*16+16)
      matvec16s<16>(lds_whh_h, lds_whh_l, WHH_LD, lds_hb_h, lds_hb_l,
                    wave, lane, lds_hg);

      // convert + store prefetched x into the slot not being read this step
      const int slot = (t & 1) * IN_SZ;
      bf16 xh0 = (bf16)xv0;
      lds_xb_h[slot + tid] = xh0;
      lds_xb_l[slot + tid] = (bf16)(xv0 - (float)xh0);
      if (tid < IN_SZ - 192) {
        bf16 xh1 = (bf16)xv1;
        lds_xb_h[slot + 192 + tid] = xh1;
        lds_xb_l[slot + 192 + tid] = (bf16)(xv1 - (float)xh1);
      }
    } else {
      // pipelined input matvec for step t+1
      const int p1 = (t + 1) & 1;
#pragma unroll
      for (int q = 0; q < 3; ++q)
        matvec16s<8>(lds_wih_h, lds_wih_l, WIH_LD,
                     lds_xb_h + p1 * IN_SZ, lds_xb_l + p1 * IN_SZ,
                     (wave - 6) * 3 + q, lane, lds_ig + p1 * ROWS);
    }

    __syncthreads();  // (B) hg and ig_{t+1} complete

    if (wave == 7) {  // 32 lanes -> this WG's 32 hidden units
      const int jj = lane;
      const float* ig = lds_ig + (t & 1) * ROWS;
      float igr = ig[jj]           + lds_bias[jj];
      float igz = ig[HPW + jj]     + lds_bias[HPW + jj];
      float ign = ig[2 * HPW + jj] + lds_bias[2 * HPW + jj];
      float rg = sigmoid_(igr + lds_hg[jj]);
      float zg = sigmoid_(igz + lds_hg[HPW + jj]);
      float ng = tanhf(ign + rg * (lds_hg[2 * HPW + jj] + lds_bn[jj]));
      float hold = lds_h[j0 + jj];
      float hnew = ng + zg * (hold - ng);
      out[(size_t)HID + (size_t)t * HID + j0 + jj] = hnew;      // out_seq[t]
      if (t == T_STEPS - 1) out[j0 + jj] = hnew;                // final_state
      __hip_atomic_store(&gh[j0 + jj], hnew, __ATOMIC_RELAXED, __HIP_MEMORY_SCOPE_AGENT);
      __threadfence();  // make h_t visible device-wide before signaling
    }

    __syncthreads();  // (C) gating + fence done

    if (in_cluster) {
      // one wave signals per workgroup; all waves wait (ISA-recommended pattern)
      if (wave == 0) asm volatile("s_barrier_signal -3" ::: "memory");
      asm volatile("s_barrier_wait -3" ::: "memory");
    } else {
      // fallback: monotonic global-atomic barrier among the 16 persistent WGs
      if (tid == 0) {
        __hip_atomic_fetch_add(bar, 1u, __ATOMIC_ACQ_REL, __HIP_MEMORY_SCOPE_AGENT);
        const unsigned target = (unsigned)(t + 1) * NWG;
        while (__hip_atomic_load(bar, __ATOMIC_ACQUIRE, __HIP_MEMORY_SCOPE_AGENT) < target) {
          __builtin_amdgcn_s_sleep(1);
        }
      }
      __syncthreads();
    }

    // gather the full updated h_t from all WGs (L2, agent scope), split hi/lo
    for (int i = tid; i < HID; i += BLOCK) {
      float v = __hip_atomic_load(&gh[i], __ATOMIC_RELAXED, __HIP_MEMORY_SCOPE_AGENT);
      bf16 vh = (bf16)v;
      lds_h[i]    = v;
      lds_hb_h[i] = vh;
      lds_hb_l[i] = (bf16)(v - (float)vh);
    }
  }
}

extern "C" void kernel_launch(void* const* d_in, const int* in_sizes, int n_in,
                              void* d_out, int out_size, void* d_ws, size_t ws_size,
                              hipStream_t stream) {
  const float* xs     = (const float*)d_in[0];
  const float* w_ih   = (const float*)d_in[1];
  const float* w_hh   = (const float*)d_in[2];
  const float* bias   = (const float*)d_in[3];
  const float* bias_n = (const float*)d_in[4];
  float* out = (float*)d_out;

  unsigned* bar = (unsigned*)d_ws;
  float*    gh  = (float*)((char*)d_ws + 256);

  const size_t smem =
      (size_t)(2 * ROWS * WHH_LD + 2 * ROWS * WIH_LD + 2 * HID + 4 * IN_SZ) * sizeof(bf16) +
      (size_t)(HID + ROWS + 2 * ROWS + ROWS + HPW) * sizeof(float);

  (void)hipFuncSetAttribute((const void*)gru_persistent,
                            hipFuncAttributeMaxDynamicSharedMemorySize, (int)smem);

  gru_init<<<1, HID, 0, stream>>>(bar, gh);
  gru_persistent<<<NWG, BLOCK, smem, stream>>>(xs, w_ih, w_hh, bias, bias_n, out, gh, bar);
}
